// DynamicSAModule_12060268167712
// MI455X (gfx1250) — compile-verified, matrix operand order checked by reading, not run on hardware
//
#include <hip/hip_runtime.h>
#include <hip/hip_bf16.h>

// ---------------- problem constants ----------------
#define BATCH   4
#define NPTS    16384
#define STRIDE  4
#define KNBR    20
#define NSCAL   16
#define NVEC    16
#define DIM     64          // NSCAL + 3*NVEC
#define HID     128
#define OUTD    128
#define TWO_D   128         // 2*DIM
#define MQ      (NPTS/STRIDE)   // 4096 queries per batch
#define NTOT    (BATCH*NPTS)    // 65536
#define NQ      (BATCH*MQ)      // 16384 destinations
#define DK      67              // DIM + 3 (pos)
#define DKP     68              // padded to multiple of 4 for 16x16x4 WMMA
#define QWAVES  4               // waves per KNN block (queries = 16*QWAVES)

typedef __attribute__((ext_vector_type(2)))  float     v2f;
typedef __attribute__((ext_vector_type(8)))  float     v8f;
typedef __attribute__((ext_vector_type(8)))  _Float16  v8h;
typedef __attribute__((ext_vector_type(16))) _Float16  v16h;

// ---- CDNA5 async global->LDS copy (ASYNCcnt-tracked DMA path) ----
__device__ __forceinline__ void async_copy_b64(unsigned int lds_addr,
                                               const float* src) {
    asm volatile("global_load_async_to_lds_b64 %0, %1, off"
                 :: "v"(lds_addr), "v"(src)
                 : "memory");
}
__device__ __forceinline__ void wait_asynccnt0() {
    asm volatile("s_wait_asynccnt 0x0" ::: "memory");
}

// ---------------------------------------------------------------------------
// Kernel 0: convert W1 [128x128] and W2 [128x128] to fp16, transposed so a
// WMMA B-tile column is contiguous:  WT[col][k] = W[k][col].
// ---------------------------------------------------------------------------
__global__ void wconv_kernel(const float* __restrict__ W1,
                             const float* __restrict__ W2,
                             _Float16* __restrict__ W1T,
                             _Float16* __restrict__ W2T) {
    int i = blockIdx.x * blockDim.x + threadIdx.x;
    if (i < TWO_D * HID) {
        int r = i / HID, c = i % HID;
        W1T[c * TWO_D + r] = (_Float16)W1[i];
    }
    if (i < HID * OUTD) {
        int r = i / OUTD, c = i % OUTD;
        W2T[c * HID + r] = (_Float16)W2[i];
    }
}

// ---------------------------------------------------------------------------
// Kernel 1: per-node key features kf = [scalars, R^T v (16 vecs), pos, 0pad]
// plus |kf|^2 (only the key part of the distance score matters for top-k).
// ---------------------------------------------------------------------------
__global__ void prep_kf_kernel(const float* __restrict__ x,
                               const float* __restrict__ pos,
                               const float* __restrict__ lf,
                               float* __restrict__ kf,
                               float* __restrict__ sq) {
    int n = blockIdx.x * blockDim.x + threadIdx.x;
    if (n >= NTOT) return;
    const float* xr = x + (size_t)n * DIM;
    const float* R  = lf + (size_t)n * 9;     // row-major 3x3
    float* o = kf + (size_t)n * DKP;
    float s2 = 0.f;
#pragma unroll
    for (int i = 0; i < NSCAL; i++) { float v = xr[i]; o[i] = v; s2 += v * v; }
#pragma unroll
    for (int k = 0; k < NVEC; k++) {
        int i0 = NSCAL + 3 * k;
        float v0 = xr[i0], v1 = xr[i0 + 1], v2 = xr[i0 + 2];
        // out = R^T v  (kf uses lframes transposed)
        float w0 = R[0] * v0 + R[3] * v1 + R[6] * v2;
        float w1 = R[1] * v0 + R[4] * v1 + R[7] * v2;
        float w2 = R[2] * v0 + R[5] * v1 + R[8] * v2;
        o[i0] = w0; o[i0 + 1] = w1; o[i0 + 2] = w2;
        s2 += w0 * w0 + w1 * w1 + w2 * w2;
    }
    float p0 = pos[n * 3], p1 = pos[n * 3 + 1], p2 = pos[n * 3 + 2];
    o[64] = p0; o[65] = p1; o[66] = p2; o[67] = 0.f;
    s2 += p0 * p0 + p1 * p1 + p2 * p2;
    sq[n] = s2;
}

// ---------------------------------------------------------------------------
// Kernel 2: brute-force KNN.  Block = 4 waves = 64 queries.  Each contiguous
// 16-key tile (16 x 68 f32 = 4352 B, rows are consecutive nodes -> one flat
// block) is DMA'd into LDS once per block via GLOBAL_LOAD_ASYNC_TO_LDS_B64
// (double-buffered, ASYNCcnt), then consumed by all 4 waves as WMMA B-tiles
// through ds_load_b64.  Distance tile via 17x V_WMMA_F32_16X16X4_F32.
// score[m][n] = |k_n|^2 - 2 q_m.k_n   (qsq constant per row -> dropped).
// ---------------------------------------------------------------------------
__launch_bounds__(32 * QWAVES)
__global__ void knn_kernel(const float* __restrict__ kf,
                           const float* __restrict__ sq,
                           int* __restrict__ knn) {
    __shared__ __align__(16) float kbuf[2][16 * DKP];   // double-buffered keys
    __shared__ float d2s[QWAVES][16][17];               // per-wave score tile

    const int blk  = blockIdx.x;                 // 0 .. NQ/(16*QWAVES)-1
    const int bpb  = MQ / (16 * QWAVES);         // blocks per batch (64)
    const int b    = blk / bpb;
    const int t64  = blk % bpb;
    const int tid  = threadIdx.x;
    const int w    = tid >> 5;
    const int lane = tid & 31;
    const int col  = lane & 15;
    const int half = lane >> 4;

    // --- load A (this wave's 16x68 query tile): lane row M=col, K-half=half
    const int qm    = t64 * (16 * QWAVES) + w * 16 + col;  // query in batch
    const int qnode = b * NPTS + qm * STRIDE;
    v2f A[17];
    {
        const float* qr = kf + (size_t)qnode * DKP + 2 * half;
#pragma unroll
        for (int c = 0; c < 17; c++) A[c] = *(const v2f*)(qr + 4 * c);
    }

    float dlist[KNBR]; int ilist[KNBR];
#pragma unroll
    for (int i = 0; i < KNBR; i++) { dlist[i] = 3.4e38f; ilist[i] = 0; }
    float worst = 3.4e38f;

    const unsigned int lds0 = (unsigned int)(uintptr_t)&kbuf[0][0];
    const unsigned int lds1 = (unsigned int)(uintptr_t)&kbuf[1][0];

    // --- prologue: wave 0 DMAs tile 0 (17 b64 per lane = 544 = 4352 B)
    if (w == 0) {
        const float* g0 = kf + ((size_t)b * NPTS) * DKP;
#pragma unroll
        for (int c = 0; c < 17; c++) {
            int i = lane + 32 * c;
            async_copy_b64(lds0 + i * 8, g0 + i * 2);
        }
        wait_asynccnt0();
    }
    __syncthreads();

    const int NT = NPTS / 16;
    for (int kt = 0; kt < NT; kt++) {
        const int buf = kt & 1;

        // prefetch next tile into the other buffer (wave 0 only)
        if (w == 0 && kt + 1 < NT) {
            const float* gs = kf + ((size_t)(b * NPTS + (kt + 1) * 16)) * DKP;
            const unsigned int lb = buf ? lds0 : lds1;
#pragma unroll
            for (int c = 0; c < 17; c++) {
                int i = lane + 32 * c;
                async_copy_b64(lb + i * 8, gs + i * 2);
            }
        }

        // --- 16x16 distance tile: B fragments straight from LDS
        v8f acc = {};
#pragma unroll
        for (int c = 0; c < 17; c++) {
            v2f Bv = *(const v2f*)&kbuf[buf][col * DKP + 4 * c + 2 * half];
            acc = __builtin_amdgcn_wmma_f32_16x16x4_f32(
                false, A[c], false, Bv, (short)0, acc, false, false);
        }

        float ks = sq[b * NPTS + kt * 16 + col];   // |key|^2 for this column
        // C layout: lane holds col, rows m = 8*half + j  (same-wave DS is
        // in-order, so no barrier needed between store and read below)
#pragma unroll
        for (int j = 0; j < 8; j++)
            d2s[w][8 * half + j][col] = ks - 2.0f * acc[j];

        if (lane < 16) {
            for (int nn = 0; nn < 16; nn++) {
                float d = d2s[w][lane][nn];
                if (d < worst) {
                    int gi = kt * 16 + nn;
                    int p = KNBR - 1;
                    while (p > 0 && dlist[p - 1] > d) {
                        dlist[p] = dlist[p - 1]; ilist[p] = ilist[p - 1]; p--;
                    }
                    dlist[p] = d; ilist[p] = gi;
                    worst = dlist[KNBR - 1];
                }
            }
        }

        // publish prefetched buffer / protect kbuf reuse
        if (w == 0) wait_asynccnt0();
        __syncthreads();
    }

    if (lane < 16) {
        int q = b * MQ + qm;        // qm already uses col == lane here
#pragma unroll
        for (int i = 0; i < KNBR; i++)
            knn[q * KNBR + i] = b * NPTS + ilist[i];
    }
}

// ---------------------------------------------------------------------------
// Kernel 3: per-destination edge MLP + max pool.
// Block = 1 destination, 256 threads = 8 waves; wave w owns output columns
// [16w, 16w+16).  H is 32x128 fp16 in LDS (rows 20..31 zero).
// Both layers via V_WMMA_F32_16X16X32_F16, fp32 accumulate.
// ---------------------------------------------------------------------------
__launch_bounds__(256)
__global__ void mlp_kernel(const float* __restrict__ x,
                           const float* __restrict__ lf,
                           const int* __restrict__ knn,
                           const _Float16* __restrict__ W1T,
                           const float* __restrict__ b1,
                           const _Float16* __restrict__ W2T,
                           const float* __restrict__ b2,
                           float* __restrict__ out) {
    __shared__ __align__(16) _Float16 Hs[32][TWO_D];
    __shared__ __align__(16) _Float16 H2s[32][HID];

    const int q = blockIdx.x;
    const int b = q / MQ, m = q % MQ;
    const int dstnode = b * NPTS + m * STRIDE;
    const int tid = threadIdx.x;

    // ---- phase 1: build H rows (edges) ----
    if (tid < KNBR) {
        int src = knn[q * KNBR + tid];
        const float* xd = x + (size_t)dstnode * DIM;
        const float* xs = x + (size_t)src * DIM;
        const float* R  = lf + (size_t)dstnode * 9;
        _Float16* row = &Hs[tid][0];
#pragma unroll
        for (int i = 0; i < NSCAL; i++) {
            float xv = xd[i];
            row[i]       = (_Float16)xv;
            row[64 + i]  = (_Float16)(xs[i] - xv);
        }
#pragma unroll
        for (int k = 0; k < NVEC; k++) {
            int i0 = NSCAL + 3 * k;
            float d0 = xd[i0], d1 = xd[i0 + 1], d2v = xd[i0 + 2];
            row[i0]     = (_Float16)d0;
            row[i0 + 1] = (_Float16)d1;
            row[i0 + 2] = (_Float16)d2v;
            float r0 = xs[i0] - d0, r1 = xs[i0 + 1] - d1, r2 = xs[i0 + 2] - d2v;
            // rel_local = R_dst @ rel   (no transpose on the edge transform)
            row[64 + i0]     = (_Float16)(R[0] * r0 + R[1] * r1 + R[2] * r2);
            row[64 + i0 + 1] = (_Float16)(R[3] * r0 + R[4] * r1 + R[5] * r2);
            row[64 + i0 + 2] = (_Float16)(R[6] * r0 + R[7] * r1 + R[8] * r2);
        }
    } else if (tid < 32) {
#pragma unroll
        for (int i = 0; i < TWO_D; i++) Hs[tid][i] = (_Float16)0.0f;
    }
    __syncthreads();

    const int w    = tid >> 5;          // wave / N-tile id
    const int lane = tid & 31;
    const int ncol = (w << 4) + (lane & 15);
    const int half = lane >> 4;
    const int arow = lane & 15;

    // ---- phase 2: layer 1 (H @ W1 + b1, relu) -> H2 ----
    {
        float bias = b1[ncol];
        for (int mt = 0; mt < 2; mt++) {
            v8f acc;
#pragma unroll
            for (int j = 0; j < 8; j++) acc[j] = bias;
#pragma unroll
            for (int kc = 0; kc < 4; kc++) {
                int r  = (mt << 4) + arow;
                int k0 = kc * 32 + 8 * half;
                v8h alo = *(const v8h*)&Hs[r][k0];
                v8h ahi = *(const v8h*)&Hs[r][k0 + 16];
                v16h av = __builtin_shufflevector(alo, ahi,
                    0,1,2,3,4,5,6,7,8,9,10,11,12,13,14,15);
                v8h blo = *(const v8h*)&W1T[(size_t)ncol * TWO_D + k0];
                v8h bhi = *(const v8h*)&W1T[(size_t)ncol * TWO_D + k0 + 16];
                v16h bv = __builtin_shufflevector(blo, bhi,
                    0,1,2,3,4,5,6,7,8,9,10,11,12,13,14,15);
                acc = __builtin_amdgcn_wmma_f32_16x16x32_f16(
                    false, av, false, bv, (short)0, acc, false, false);
            }
#pragma unroll
            for (int j = 0; j < 8; j++) {
                float v = acc[j];
                v = v > 0.f ? v : 0.f;                       // relu
                H2s[(mt << 4) + 8 * half + j][ncol] = (_Float16)v;
            }
        }
    }
    __syncthreads();

    // ---- phase 3: layer 2 (H2 @ W2 + b2), masked max over rows 0..19 ----
    float pmax = -3.4e38f;
    {
        float bias = b2[ncol];
        for (int mt = 0; mt < 2; mt++) {
            v8f acc;
#pragma unroll
            for (int j = 0; j < 8; j++) acc[j] = bias;
#pragma unroll
            for (int kc = 0; kc < 4; kc++) {
                int r  = (mt << 4) + arow;
                int k0 = kc * 32 + 8 * half;
                v8h alo = *(const v8h*)&H2s[r][k0];
                v8h ahi = *(const v8h*)&H2s[r][k0 + 16];
                v16h av = __builtin_shufflevector(alo, ahi,
                    0,1,2,3,4,5,6,7,8,9,10,11,12,13,14,15);
                v8h blo = *(const v8h*)&W2T[(size_t)ncol * HID + k0];
                v8h bhi = *(const v8h*)&W2T[(size_t)ncol * HID + k0 + 16];
                v16h bv = __builtin_shufflevector(blo, bhi,
                    0,1,2,3,4,5,6,7,8,9,10,11,12,13,14,15);
                acc = __builtin_amdgcn_wmma_f32_16x16x32_f16(
                    false, av, false, bv, (short)0, acc, false, false);
            }
            if (mt == 0) {          // rows 0..15 all valid
#pragma unroll
                for (int j = 0; j < 8; j++) pmax = fmaxf(pmax, acc[j]);
            } else if (half == 0) { // rows 16..19 valid only (j<4, half 0)
#pragma unroll
                for (int j = 0; j < 4; j++) pmax = fmaxf(pmax, acc[j]);
            }
        }
    }
    // combine the two half-lanes holding the same column
    float other = __shfl_xor(pmax, 16, 32);
    pmax = fmaxf(pmax, other);
    if (half == 0) out[(size_t)q * OUTD + ncol] = pmax;
}

// ---------------------------------------------------------------------------
// Kernel 4: aux tuple outputs: pos[idx], batch[idx] (as float), lf_dst.
// ---------------------------------------------------------------------------
__global__ void aux_kernel(const float* __restrict__ pos,
                           const int* __restrict__ batch,
                           const float* __restrict__ lf,
                           float* __restrict__ out) {
    int q = blockIdx.x * blockDim.x + threadIdx.x;
    if (q >= NQ) return;
    int b = q / MQ, m = q % MQ;
    int node = b * NPTS + m * STRIDE;
    float* po = out + (size_t)NQ * OUTD;
    po[q * 3 + 0] = pos[node * 3 + 0];
    po[q * 3 + 1] = pos[node * 3 + 1];
    po[q * 3 + 2] = pos[node * 3 + 2];
    float* bo = po + (size_t)NQ * 3;
    bo[q] = (float)batch[node];
    float* lo = bo + NQ;
#pragma unroll
    for (int i = 0; i < 9; i++) lo[q * 9 + i] = lf[(size_t)node * 9 + i];
}

// ---------------------------------------------------------------------------
extern "C" void kernel_launch(void* const* d_in, const int* in_sizes, int n_in,
                              void* d_out, int out_size, void* d_ws, size_t ws_size,
                              hipStream_t stream) {
    const float* x    = (const float*)d_in[0];
    const float* pos  = (const float*)d_in[1];
    const float* lf   = (const float*)d_in[2];
    const int*   bat  = (const int*)  d_in[3];
    const float* W1   = (const float*)d_in[4];
    const float* b1   = (const float*)d_in[5];
    const float* W2   = (const float*)d_in[6];
    const float* b2   = (const float*)d_in[7];
    float* out = (float*)d_out;

    // workspace carve-up
    float* kf  = (float*)d_ws;                          // NTOT * DKP
    float* sq  = kf + (size_t)NTOT * DKP;               // NTOT
    int*   knn = (int*)(sq + NTOT);                     // NQ * KNBR
    _Float16* W1T = (_Float16*)(knn + (size_t)NQ * KNBR);   // HID * TWO_D
    _Float16* W2T = W1T + (size_t)HID * TWO_D;              // OUTD * HID
    (void)in_sizes; (void)n_in; (void)out_size; (void)ws_size;

    wconv_kernel<<<(TWO_D * HID + 255) / 256, 256, 0, stream>>>(W1, W2, W1T, W2T);
    prep_kf_kernel<<<(NTOT + 255) / 256, 256, 0, stream>>>(x, pos, lf, kf, sq);
    knn_kernel<<<NQ / (16 * QWAVES), 32 * QWAVES, 0, stream>>>(kf, sq, knn);
    mlp_kernel<<<NQ, 256, 0, stream>>>(x, lf, knn, W1T, b1, W2T, b2, out);
    aux_kernel<<<(NQ + 255) / 256, 256, 0, stream>>>(pos, bat, lf, out);
}
